// Attention_11141145166261
// MI455X (gfx1250) — matrix operand order repaired and checked by reading, not compiled, not run on hardware
//
#include <hip/hip_runtime.h>
#include <hip/hip_bf16.h>

#define HID    1024
#define TWO_H  2048
#define BATCH  64
#define SRC    512
#define NEG_VAL (-1e10f)

// clang ext-vector types (trivial, properly aligned, union-safe)
typedef __attribute__((ext_vector_type(16))) __bf16       v16bf;
typedef __attribute__((ext_vector_type(8)))  float        v8f;
typedef __attribute__((ext_vector_type(4)))  __bf16       bf16x4;
typedef __attribute__((ext_vector_type(4)))  float        f32x4;
typedef __attribute__((ext_vector_type(4)))  unsigned int u32x4;

// ---------------------------------------------------------------------------
// Kernel 1: Wh[b][h] = sum_k hidden[b,k] * W_w[h,k] + W_b[h]     (64 x 1024)
// Tiny (134 MFLOP); W_w (4MB) and hidden (256KB) are L2 resident.
// ---------------------------------------------------------------------------
__global__ __launch_bounds__(256)
void wh_kernel(const float* __restrict__ hidden,
               const float* __restrict__ W_w,
               const float* __restrict__ W_b,
               float* __restrict__ Wh) {
  const int g = blockIdx.x * blockDim.x + threadIdx.x;   // 65536 threads
  const int b = g >> 10;
  const int h = g & 1023;
  const f32x4* hr = (const f32x4*)(hidden + (size_t)b * HID);
  const f32x4* wr = (const f32x4*)(W_w    + (size_t)h * HID);
  float acc = 0.f;
  for (int k = 0; k < HID / 4; ++k) {
    f32x4 x = hr[k], w = wr[k];
    acc += x.x * w.x + x.y * w.y + x.z * w.z + x.w * w.w;
  }
  Wh[g] = acc + W_b[h];
}

// ---------------------------------------------------------------------------
// Kernel 2: one-time convert U_w (f32 [H][2H]) -> bf16 workspace, same layout.
// Natural [n][k] layout == contiguous-K per fixed N, which is exactly the
// per-lane access the WMMA B-fragment wants.
// ---------------------------------------------------------------------------
__global__ __launch_bounds__(256)
void cvt_uw_kernel(const float* __restrict__ U_w, __bf16* __restrict__ Ubf) {
  const int g = blockIdx.x * blockDim.x + threadIdx.x;   // HID*TWO_H/4 threads
  f32x4 f = ((const f32x4*)U_w)[g];
  bf16x4 o = { (__bf16)f.x, (__bf16)f.y, (__bf16)f.z, (__bf16)f.w };
  ((bf16x4*)Ubf)[g] = o;
}

// ---------------------------------------------------------------------------
// Kernel 3: fused energy kernel.
//   GEMM rows r = s*64 + b  (memory order of encoder_outputs [S,B,2H])
//   acc[r][n] = sum_k enc[r,k] * U_w[n,k]      (bf16 WMMA, f32 accumulate)
//   e[r]      = sum_n tanh(acc + Wh[b,n] + U_b[n]) * v[n]
//   energy[b][s] = mask[b][s] ? NEG : e[r]     (pre-softmax, to d_out)
// ---------------------------------------------------------------------------
#define BM 32                 // rows per workgroup
#define AP (TWO_H + 8)        // padded LDS row stride (elements) -> no bank conflicts

__global__ __launch_bounds__(512)
void energy_kernel(const float* __restrict__ enc,        // [S*B, 2H]
                   const __bf16* __restrict__ Ubf,       // [H][2H] bf16
                   const float* __restrict__ Wh,         // [B][H]
                   const float* __restrict__ U_b,        // [H]
                   const float* __restrict__ v_w,        // [H]
                   const unsigned char* __restrict__ mask, // [B][S] (bool)
                   float* __restrict__ energy)           // [B][S]
{
  __shared__ __bf16 sA[BM * AP];
  __shared__ float  sE[BM];

  const int tid = threadIdx.x;
  const int R0  = blockIdx.x * BM;

  if (tid < BM) sE[tid] = 0.f;

  // ---- stage A tile: 32 rows x 2048, f32 -> bf16 into padded LDS ----
  for (int idx = tid; idx < BM * (TWO_H / 4); idx += 512) {
    const int m  = idx >> 9;          // idx / 512
    const int k4 = idx & 511;         // idx % 512
    f32x4 f = ((const f32x4*)enc)[(size_t)(R0 + m) * (TWO_H / 4) + k4];
    bf16x4 o = { (__bf16)f.x, (__bf16)f.y, (__bf16)f.z, (__bf16)f.w };
    *(bf16x4*)(&sA[m * AP + k4 * 4]) = o;
  }
  __syncthreads();

  // ---- wave tiling: 16 waves = 2 row-groups x 8 col-groups ----
  const int lane = tid & 31;
  const int wave = tid >> 5;
  const int m0   = (wave >> 3) * 16;    // 0 or 16
  const int n0   = (wave & 7) * 128;    // 0..896
  const int lm   = lane & 15;
  const int half = lane >> 4;

  // A fragment (16-bit A 16x32): lanes<16 hold K {0..15}; lanes>=16 hold
  // K {16..31}  -> two 16B LDS reads at (half*16) and (half*16+8).
  int aoff = (m0 + lm) * AP + half * 16;

  // B fragment (16-bit B 32x16): lane needs contiguous K=half*16..+15 of
  // column n -> two 16B global reads from Ubf row n.
  const u32x4* Bq = (const u32x4*)Ubf;
  int bbase[8];
#pragma unroll
  for (int t = 0; t < 8; ++t)
    bbase[t] = ((n0 + 16 * t + lm) * TWO_H + half * 16) >> 3;  // u32x4 index

  v8f acc[8];
#pragma unroll
  for (int t = 0; t < 8; ++t) acc[t] = (v8f){0, 0, 0, 0, 0, 0, 0, 0};

  union AF { v16bf v; u32x4 q[2]; };

  for (int k0 = 0; k0 < TWO_H; k0 += 32) {
    AF a;
    a.q[0] = *(const u32x4*)(&sA[aoff]);
    a.q[1] = *(const u32x4*)(&sA[aoff + 8]);
    aoff += 32;
    const int kq = k0 >> 3;
#pragma unroll
    for (int t = 0; t < 8; ++t) {
      AF bb;
      bb.q[0] = Bq[bbase[t] + kq];
      bb.q[1] = Bq[bbase[t] + kq + 1];
      acc[t] = __builtin_amdgcn_wmma_f32_16x16x32_bf16(
          /*neg_a=*/false, a.v, /*neg_b=*/false, bb.v,
          /*c_mod=*/(short)0, acc[t], /*reuse_a=*/false, /*reuse_b=*/false);
    }
  }

  // ---- epilogue: tanh(acc + Wh + U_b) . v, reduce over N ----
  // C layout: VGPR i -> row (m0 + half*8 + i), col n = n0+16t+lm.
  float e[8] = {0, 0, 0, 0, 0, 0, 0, 0};
#pragma unroll
  for (int t = 0; t < 8; ++t) {
    const int n  = n0 + 16 * t + lm;
    const float vb = v_w[n];
    const float ub = U_b[n];
#pragma unroll
    for (int i = 0; i < 8; ++i) {
      const int m = m0 + half * 8 + i;
      const int b = (R0 + m) & (BATCH - 1);
      const float x = acc[t][i] + Wh[b * HID + n] + ub;
      e[i] += tanhf(x) * vb;
    }
  }
  // reduce across the 16 lanes sharing each row
#pragma unroll
  for (int off = 1; off < 16; off <<= 1) {
#pragma unroll
    for (int i = 0; i < 8; ++i) e[i] += __shfl_xor(e[i], off, 16);
  }
  if (lm == 0) {
#pragma unroll
    for (int i = 0; i < 8; ++i)
      atomicAdd(&sE[m0 + half * 8 + i], e[i]);
  }
  __syncthreads();

  if (tid < BM) {
    const int r = R0 + tid;
    const int b = r & (BATCH - 1);
    const int s = r >> 6;
    float val = sE[tid];
    if (mask[b * SRC + s]) val = NEG_VAL;   // reference: where(mask, NEG, e)
    energy[b * SRC + s] = val;
  }
}

// ---------------------------------------------------------------------------
// Kernel 4: row-wise softmax over [64][512], in place on d_out.
// ---------------------------------------------------------------------------
__global__ __launch_bounds__(512)
void softmax_kernel(float* __restrict__ out) {
  __shared__ float red[512];
  const int b = blockIdx.x;
  const int s = threadIdx.x;
  const float e = out[b * SRC + s];
  red[s] = e;
  __syncthreads();
  for (int off = 256; off > 0; off >>= 1) {
    if (s < off) red[s] = fmaxf(red[s], red[s + off]);
    __syncthreads();
  }
  const float mx = red[0];
  __syncthreads();
  const float ex = __expf(e - mx);
  red[s] = ex;
  __syncthreads();
  for (int off = 256; off > 0; off >>= 1) {
    if (s < off) red[s] += red[s + off];
    __syncthreads();
  }
  out[b * SRC + s] = ex / red[0];
}

// ---------------------------------------------------------------------------
extern "C" void kernel_launch(void* const* d_in, const int* in_sizes, int n_in,
                              void* d_out, int out_size, void* d_ws, size_t ws_size,
                              hipStream_t stream) {
  const float*         hidden = (const float*)d_in[0];
  const float*         enc    = (const float*)d_in[1];
  const unsigned char* mask   = (const unsigned char*)d_in[2];
  const float*         W_w    = (const float*)d_in[3];
  const float*         W_b    = (const float*)d_in[4];
  const float*         U_w    = (const float*)d_in[5];
  const float*         U_b    = (const float*)d_in[6];
  const float*         v_w    = (const float*)d_in[7];
  float* out = (float*)d_out;

  char* ws = (char*)d_ws;
  float*  Wh  = (float*)ws;                         // 64*1024*4   = 256 KB
  __bf16* Ubf = (__bf16*)(ws + 262144);             // 1024*2048*2 =   4 MB

  wh_kernel<<<(BATCH * HID) / 256, 256, 0, stream>>>(hidden, W_w, W_b, Wh);
  cvt_uw_kernel<<<(HID * TWO_H / 4) / 256, 256, 0, stream>>>(U_w, Ubf);
  energy_kernel<<<(SRC * BATCH) / BM, 512, 0, stream>>>(enc, Ubf, Wh, U_b, v_w,
                                                        mask, out);
  softmax_kernel<<<BATCH, SRC, 0, stream>>>(out);
}
